// SeaReader_v2_37091337568678
// MI455X (gfx1250) — matrix operand order repaired and checked by reading, not compiled, not run on hardware
//
#include <hip/hip_runtime.h>
#include <hip/hip_bf16.h>
#include <cstdint>

// ===================== CDNA5 (gfx1250) types =====================
typedef __attribute__((ext_vector_type(16))) _Float16 v16h;
typedef __attribute__((ext_vector_type(8)))  float    v8f;
typedef __attribute__((ext_vector_type(4)))  unsigned int u32x4;
typedef __attribute__((ext_vector_type(8)))  int      i32x8;
typedef __attribute__((ext_vector_type(4)))  int      i32x4;
typedef _Float16 h16;

__device__ __forceinline__ float sigmoidf_(float x) { return 1.0f / (1.0f + __expf(-x)); }

// -----------------------------------------------------------------
// Fragment-order LDS layout helpers (CDNA5 ISA 7.12.2, wave32):
//  A 16x32 f16 : lane = (m&15) | (khalf<<4), khalf=(c>>3)&1,
//                idx  = (c&7) | ((c>>4)<<3)          (16 halves/lane)
//  B 32x16 f16 : lane = (n&15) | (((k>>3)&1)<<4), idx = (k&7) | ((k>>4)<<3)
// Storing tiles in [tile][lane][16] order makes every fragment one
// contiguous 32-byte LDS read (2x ds_load_b128, no v_mov_b16 glue).
// -----------------------------------------------------------------

// =================================================================
// TDM: 2D tile load Global->LDS (rows x 512 dwords, row stride ldT*512
// dwords, rows beyond `rows` zero-filled via tensor_dim1 OOB rule).
// D# packed per ISA 8.3/8.4; amdgpu-toolchain 6-arg builtin.
// =================================================================
__device__ __forceinline__ void tdm_load_rows512(const float* gsrc, unsigned ldsOff,
                                                 int rows, int ldT)
{
  unsigned long long ga = (unsigned long long)(uintptr_t)gsrc;
  u32x4 g0;
  g0[0] = 1u;                                                  // count=1, user mode
  g0[1] = ldsOff;                                              // lds_addr (bytes)
  g0[2] = (unsigned)(ga & 0xFFFFFFFFu);                        // global_addr[31:0]
  g0[3] = (unsigned)((ga >> 32) & 0x01FFFFFFu) | (2u << 30);   // global_addr[56:32] | type=2
  i32x8 g1;
  g1[0] = (int)(2u << 16);                                     // wg_mask=0 | data_size=4B
  g1[1] = (int)(512u << 16);                                   // tensor_dim0=512 (lo16)
  g1[2] = (int)(((unsigned)rows & 0xFFFFu) << 16);             // dim0 hi=0 | tensor_dim1 lo
  g1[3] = (int)(512u << 16);                                   // dim1 hi=0 | tile_dim0=512
  g1[4] = 16;                                                  // tile_dim1=16, tile_dim2=0
  g1[5] = (int)((unsigned)ldT * 512u);                         // tensor_dim0_stride lo32
  g1[6] = 0;                                                   // stride hi | dim1_stride lo
  g1[7] = 0;
  i32x4 gz4 = {0, 0, 0, 0};
  i32x8 gz8 = {0, 0, 0, 0, 0, 0, 0, 0};
  __builtin_amdgcn_tensor_load_to_lds(g0, g1, gz4, gz4, gz8, 0);
}

// =================================================================
// Generic batched WMMA GEMM:  C(f32)[,Ch(f16)] = A(f16) x B(f16) (+bias)
//   BT==0 : B row-major [K x N] ;  BT==1 : B row-major [N x K] (C=A.B^T)
// Block 128 threads / 4 waves; tile 64x64; K step 32; tiles staged in
// fragment order so each WMMA operand is one 32B contiguous LDS read.
// =================================================================
__global__ __launch_bounds__(128)
void gemm_wmma(const h16* __restrict__ A, long long lda, long long strideA, int nPerA,
               const h16* __restrict__ Bm, long long ldb, long long strideB, int nPerB,
               float* __restrict__ C, long long ldc, long long strideC,
               h16* __restrict__ Ch, long long ldch, long long strideCh,
               const float* __restrict__ bias,
               int M, int N, int K, int BT)
{
  __shared__ __align__(32) h16 sA[4 * 32 * 16];   // 4 m-tiles, frag order
  __shared__ __align__(32) h16 sB[4 * 32 * 16];   // 4 n-tiles, frag order

  int z = blockIdx.z;
  A  += (long long)(z / nPerA) * strideA;
  Bm += (long long)(z / nPerB) * strideB;
  if (C)  C  += (long long)z * strideC;
  if (Ch) Ch += (long long)z * strideCh;

  int m0 = blockIdx.x * 64, n0 = blockIdx.y * 64;
  int tid = threadIdx.x;
  int wave = tid >> 5, lane = tid & 31;
  int wm = (wave >> 1) * 32, wn = (wave & 1) * 32;

  v8f acc[2][2] = {};

  for (int k0 = 0; k0 < K; k0 += 32) {
    // stage A tile [64 x 32] into fragment order
    for (int i = tid; i < 64 * 32; i += 128) {
      int r = i >> 5, c = i & 31;
      int gm = m0 + r, gk = k0 + c;
      h16 v = (h16)0.0f;
      if (gm < M && gk < K) v = A[(long long)gm * lda + gk];
      int lane_ = (r & 15) | (((c >> 3) & 1) << 4);
      int idx   = (c & 7) | ((c >> 4) << 3);
      sA[(((r >> 4) * 32 + lane_) << 4) + idx] = v;
    }
    // stage B tile [32 x 64] into fragment order
    for (int i = tid; i < 32 * 64; i += 128) {
      int kk = i >> 6, nn = i & 63;
      int gk = k0 + kk, gn = n0 + nn;
      h16 v = (h16)0.0f;
      if (gk < K && gn < N)
        v = BT ? Bm[(long long)gn * ldb + gk] : Bm[(long long)gk * ldb + gn];
      int lane_ = (nn & 15) | (((kk >> 3) & 1) << 4);
      sB[(((nn >> 4) * 32 + lane_) << 4) + ((kk & 7) | ((kk >> 4) << 3))] = v;
    }
    // prefetch next K tile of A (global_prefetch on gfx1250)
    if (k0 + 32 < K && tid < 64 && (m0 + tid) < M)
      __builtin_prefetch(A + (long long)(m0 + tid) * lda + (k0 + 32));
    __syncthreads();

    v16h af[2], bf[2];
    #pragma unroll
    for (int t = 0; t < 2; ++t) {
      af[t] = *(const v16h*)(sA + ((((wm >> 4) + t) * 32 + lane) << 4));
      bf[t] = *(const v16h*)(sB + ((((wn >> 4) + t) * 32 + lane) << 4));
    }
    #pragma unroll
    for (int i = 0; i < 2; ++i)
      #pragma unroll
      for (int j = 0; j < 2; ++j)
        acc[i][j] = __builtin_amdgcn_wmma_f32_16x16x32_f16(
            false, af[i], false, bf[j], (short)0, acc[i][j], false, false);
    __syncthreads();
  }

  // epilogue (C layout: lane n=l%16, mb=(l/16)*8, VGPR r -> M=mb+r)
  int cn = lane & 15, cmb = (lane >> 4) * 8;
  #pragma unroll
  for (int i = 0; i < 2; ++i)
    #pragma unroll
    for (int j = 0; j < 2; ++j) {
      int gn = n0 + wn + j * 16 + cn;
      if (gn >= N) continue;
      float bv = bias ? bias[gn] : 0.0f;
      #pragma unroll
      for (int r = 0; r < 8; ++r) {
        int gm = m0 + wm + i * 16 + cmb + r;
        if (gm >= M) continue;
        float val = acc[i][j][r] + bv;
        if (C)  C [(long long)gm * ldc  + gn] = val;
        if (Ch) Ch[(long long)gm * ldch + gn] = (h16)val;
      }
    }
}

// =================================================================
// Persistent LSTM scan (one direction per call).
// xg : precomputed x@Wi + b, f32 [S, T, 512] (gate order i,f,g,o)
// Wh : f16 [128,512], reordered once into fragment-order LDS (128KB).
// hs : f32 [S, T, hsld], hidden written at column colOff.
// Block = 128 thr / 4 waves = 16 sequences; wave w owns gate w's 128
// columns. h kept f16 in LDS in fragment order; c f32 in LDS.
// Per-step xg slab (16x512 f32) is double-buffered in LDS and fetched
// one step ahead by the Tensor Data Mover (wave 0), synchronized with
// s_wait_tensorcnt + workgroup barrier.
// LDS: Whf 131072 | Hf 4096 | C 8192 | G 32768 | X 2x32768 = 241664 B
// =================================================================
#define LSTM_SHMEM 241664

__global__ __launch_bounds__(128)
void lstm_scan(const float* __restrict__ xg, const h16* __restrict__ Wh,
               float* __restrict__ hs, int hsld, int colOff,
               int S, int T, int rev)
{
  extern __shared__ __align__(32) char smem[];
  h16*   sWhf = (h16*)smem;                            // frag order [kc][ntile][lane][16]
  h16*   sHf  = (h16*)(smem + 131072);                 // frag order [kc][lane][16]
  float* sC   = (float*)(smem + 131072 + 4096);        // [16][128]
  float* sG   = (float*)(smem + 131072 + 4096 + 8192); // [16][512]
  float* sX   = (float*)(smem + 176128);               // [2][16][512]

  int tid = threadIdx.x, wave = tid >> 5, lane = tid & 31;
  int s0 = blockIdx.x * 16;
  int rowsAvail = S - s0; if (rowsAvail > 16) rowsAvail = 16;

  // reorder Wh[128,512] into fragment order: k=(kc*32+kk), n=(ntile*16+nl)
  for (int i = tid; i < 128 * 512; i += 128) {
    int k = i >> 9, n = i & 511;
    int kc = k >> 5, kk = k & 31;
    int lane_ = (n & 15) | (((kk >> 3) & 1) << 4);
    sWhf[((((kc << 5) + (n >> 4)) * 32 + lane_) << 4) + ((kk & 7) | ((kk >> 4) << 3))] = Wh[i];
  }
  for (int i = tid; i < 4 * 32 * 16; i += 128) sHf[i] = (h16)0.0f;
  for (int i = tid; i < 16 * 128; i += 128) sC[i] = 0.0f;

  // TDM: fetch step-0 xg slab into buffer 0 (wave 0 only; EXEC-independent)
  if (wave == 0) {
    int t0 = rev ? (T - 1) : 0;
    tdm_load_rows512(xg + ((long long)s0 * T + t0) * 512,
                     (unsigned)(uintptr_t)sX, rowsAvail, T);
  }
  __syncthreads();

  for (int step = 0; step < T; ++step) {
    int t = rev ? (T - 1 - step) : step;

    // ---- h @ Wh for this wave's 128 gate columns ----
    v8f acc[8] = {};
    #pragma unroll
    for (int kc = 0; kc < 4; ++kc) {
      v16h a = *(const v16h*)(sHf + ((kc * 32 + lane) << 4));
      #pragma unroll
      for (int j = 0; j < 8; ++j) {
        v16h bfr = *(const v16h*)(sWhf + ((((kc << 5) + (wave * 8 + j)) * 32 + lane) << 4));
        acc[j] = __builtin_amdgcn_wmma_f32_16x16x32_f16(
            false, a, false, bfr, (short)0, acc[j], false, false);
      }
    }
    // spill raw gate pre-activations
    {
      int cn = lane & 15, cmb = (lane >> 4) * 8;
      #pragma unroll
      for (int j = 0; j < 8; ++j)
        #pragma unroll
        for (int r = 0; r < 8; ++r)
          sG[(cmb + r) * 512 + wave * 128 + j * 16 + cn] = acc[j][r];
    }
    // ---- TDM prefetch next xg slab; ensure current slab landed ----
    if (wave == 0) {
      if (step + 1 < T) {
        int tn = rev ? (T - 2 - step) : (step + 1);
        tdm_load_rows512(xg + ((long long)s0 * T + tn) * 512,
                         (unsigned)(uintptr_t)(sX + ((step + 1) & 1) * 8192),
                         rowsAvail, T);
        __builtin_amdgcn_s_wait_tensorcnt((short)1);
      } else {
        __builtin_amdgcn_s_wait_tensorcnt((short)0);
      }
    }
    __syncthreads();

    // ---- elementwise gate math; h written back in fragment order ----
    const float* xb = sX + (step & 1) * 8192;
    for (int e = tid; e < 2048; e += 128) {
      int m = e >> 7, n = e & 127;
      int s = s0 + m;
      const float* xr = xb + m * 512;
      float iv = sG[m * 512 + n]       + xr[n];
      float fv = sG[m * 512 + 128 + n] + xr[128 + n];
      float gv = sG[m * 512 + 256 + n] + xr[256 + n];
      float ov = sG[m * 512 + 384 + n] + xr[384 + n];
      float c = sigmoidf_(fv) * sC[m * 128 + n] + sigmoidf_(iv) * tanhf(gv);
      float h = sigmoidf_(ov) * tanhf(c);
      sC[m * 128 + n] = c;
      int kc = n >> 5, cc = n & 31;
      int lane_ = (m & 15) | (((cc >> 3) & 1) << 4);
      int idx   = (cc & 7) | ((cc >> 4) << 3);
      sHf[((kc * 32 + lane_) << 4) + idx] = (h16)h;
      if (s < S) hs[((long long)s * T + t) * hsld + colOff + n] = h;
    }
    __syncthreads();
  }
}

// ============ gather embeddings (fixed + delta) -> padded f16 ============
__global__ void gather_embed(const int* __restrict__ tok, long long ntok,
                             const float* __restrict__ ef, const float* __restrict__ ed,
                             h16* __restrict__ out, int E_, int Kp)
{
  long long total = ntok * Kp;
  for (long long idx = (long long)blockIdx.x * blockDim.x + threadIdx.x; idx < total;
       idx += (long long)gridDim.x * blockDim.x) {
    long long t = idx / Kp; int e = (int)(idx % Kp);
    float v = 0.0f;
    if (e < E_) { long long o = (long long)tok[t] * E_ + e; v = ef[o] + ed[o]; }
    out[idx] = (h16)v;
  }
}

// ============ f32 -> f16 2D copy-convert with zero row padding ============
__global__ void convert2d(const float* __restrict__ src, long long sld,
                          h16* __restrict__ dst, long long dld,
                          int srcRows, int dstRows, int cols)
{
  long long total = (long long)dstRows * cols;
  for (long long idx = (long long)blockIdx.x * blockDim.x + threadIdx.x; idx < total;
       idx += (long long)gridDim.x * blockDim.x) {
    long long r = idx / cols; int c = (int)(idx % cols);
    float v = (r < srcRows) ? src[r * sld + c] : 0.0f;
    dst[r * dld + c] = (h16)v;
  }
}

// ============ row softmax (+optional row max/mean features), f16 out ============
__global__ void softmax_row(const float* __restrict__ in, h16* __restrict__ out,
                            float* __restrict__ feat, int len)
{
  long long row = blockIdx.x;
  const float* r = in + row * (long long)len;
  h16* o = out + row * (long long)len;
  __shared__ float red[256];
  int tid = threadIdx.x, bs = blockDim.x;
  float mx = -3.402823e38f, rs = 0.0f;
  for (int i = tid; i < len; i += bs) { float v = r[i]; mx = fmaxf(mx, v); rs += v; }
  red[tid] = mx; __syncthreads();
  for (int s = bs >> 1; s > 0; s >>= 1) { if (tid < s) red[tid] = fmaxf(red[tid], red[tid + s]); __syncthreads(); }
  mx = red[0]; __syncthreads();
  red[tid] = rs; __syncthreads();
  for (int s = bs >> 1; s > 0; s >>= 1) { if (tid < s) red[tid] += red[tid + s]; __syncthreads(); }
  rs = red[0]; __syncthreads();
  float es = 0.0f;
  for (int i = tid; i < len; i += bs) es += __expf(r[i] - mx);
  red[tid] = es; __syncthreads();
  for (int s = bs >> 1; s > 0; s >>= 1) { if (tid < s) red[tid] += red[tid + s]; __syncthreads(); }
  float inv = 1.0f / red[0];
  for (int i = tid; i < len; i += bs) o[i] = (h16)(__expf(r[i] - mx) * inv);
  if (feat && tid == 0) { feat[row * 2 + 0] = mx; feat[row * 2 + 1] = rs / (float)len; }
}

// ====== column softmax over q (stride L), writes TRANSPOSED f16 [z,l,q] ======
__global__ void softmax_col(const float* __restrict__ Mq, h16* __restrict__ outT,
                            float* __restrict__ feat, int Q_, int L_)
{
  long long zl = blockIdx.x;
  long long z = zl / L_; int l = (int)(zl % L_);
  const float* base = Mq + z * (long long)Q_ * L_ + l;
  h16* o = outT + z * (long long)L_ * Q_ + (long long)l * Q_;
  __shared__ float red[128];
  int tid = threadIdx.x, bs = blockDim.x;
  float mx = -3.402823e38f, rs = 0.0f;
  for (int q = tid; q < Q_; q += bs) { float v = base[(long long)q * L_]; mx = fmaxf(mx, v); rs += v; }
  red[tid] = mx; __syncthreads();
  for (int s = bs >> 1; s > 0; s >>= 1) { if (tid < s) red[tid] = fmaxf(red[tid], red[tid + s]); __syncthreads(); }
  mx = red[0]; __syncthreads();
  red[tid] = rs; __syncthreads();
  for (int s = bs >> 1; s > 0; s >>= 1) { if (tid < s) red[tid] += red[tid + s]; __syncthreads(); }
  rs = red[0]; __syncthreads();
  float es = 0.0f;
  for (int q = tid; q < Q_; q += bs) es += __expf(base[(long long)q * L_] - mx);
  red[tid] = es; __syncthreads();
  for (int s = bs >> 1; s > 0; s >>= 1) { if (tid < s) red[tid] += red[tid + s]; __syncthreads(); }
  float inv = 1.0f / red[0];
  for (int q = tid; q < Q_; q += bs) o[q] = (h16)(__expf(base[(long long)q * L_] - mx) * inv);
  if (feat && tid == 0) { feat[zl * 2 + 0] = mx; feat[zl * 2 + 1] = rs / (float)Q_; }
}

// ============ sigmoid(X[row,:] . W + b) + eps ============
__global__ void rowdot_sigmoid(const float* __restrict__ X, int ld,
                               const float* __restrict__ W, const float* __restrict__ bb,
                               float* __restrict__ out, int Kd, float eps)
{
  long long row = blockIdx.x;
  __shared__ float red[128];
  int tid = threadIdx.x;
  float s = 0.0f;
  for (int i = tid; i < Kd; i += 128) s += X[row * (long long)ld + i] * W[i];
  red[tid] = s; __syncthreads();
  for (int st = 64; st > 0; st >>= 1) { if (tid < st) red[tid] += red[tid + st]; __syncthreads(); }
  if (tid == 0) out[row] = sigmoidf_(red[0] + bb[0]) + eps;
}

// ============ build second-LSTM inputs (gated concat, padded f16) ============
__global__ void build_rnq_in(const float* __restrict__ RnQ, const float* __restrict__ fc,
                             const float* __restrict__ gq, h16* __restrict__ out)
{
  const int Kp = 288;
  long long total = 100LL * 100 * Kp;
  for (long long idx = (long long)blockIdx.x * blockDim.x + threadIdx.x; idx < total;
       idx += (long long)gridDim.x * blockDim.x) {
    long long z = idx / (100 * Kp);
    int rem = (int)(idx % (100 * Kp));
    int q = rem / Kp, k = rem % Kp;
    int b = (int)(z / 10);
    float g = gq[b * 100 + q];
    long long rq = z * 100 + q;
    float v = 0.0f;
    if (k < 256) v = RnQ[rq * 256 + k];
    else if (k < 258) v = fc[rq * 2 + (k - 256)];
    out[idx] = (h16)(v * g);
  }
}

__global__ void build_rmd_in(const float* __restrict__ RmD, const float* __restrict__ fr,
                             const float* __restrict__ gc, h16* __restrict__ out)
{
  const int Kp = 544;
  long long total = 100LL * 256 * Kp;
  for (long long idx = (long long)blockIdx.x * blockDim.x + threadIdx.x; idx < total;
       idx += (long long)gridDim.x * blockDim.x) {
    long long z = idx / (256 * Kp);
    int rem = (int)(idx % (256 * Kp));
    int l = rem / Kp, k = rem % Kp;
    long long rl = z * 256 + l;
    float g = gc[rl];
    float v = 0.0f;
    if (k < 512) v = RmD[rl * 512 + k];
    else if (k < 514) v = fr[rl * 2 + (k - 512)];
    out[idx] = (h16)(v * g);
  }
}

// ============ max over time ============
__global__ void max_time(const float* __restrict__ hs, float* __restrict__ out, int T, int D)
{
  int s = blockIdx.x, j = threadIdx.x;
  float m = -3.402823e38f;
  for (int t = 0; t < T; ++t) m = fmaxf(m, hs[((long long)s * T + t) * D + j]);
  out[(long long)s * D + j] = m;
}

// ============ node gate: gf = feat * sigmoid(feat.W + b) ============
__global__ void gate_feat(const float* __restrict__ qo, const float* __restrict__ co,
                          const float* __restrict__ W, const float* __restrict__ bb,
                          float* __restrict__ gf)
{
  int z = blockIdx.x, tid = threadIdx.x;
  __shared__ float red[128];
  float s = 0.0f;
  for (int i = tid; i < 512; i += 128) {
    float f = (i < 256) ? qo[z * 256 + i] : co[z * 256 + (i - 256)];
    s += f * W[i];
  }
  red[tid] = s; __syncthreads();
  for (int st = 64; st > 0; st >>= 1) { if (tid < st) red[tid] += red[tid + st]; __syncthreads(); }
  float g = sigmoidf_(red[0] + bb[0]);
  for (int i = tid; i < 512; i += 128) {
    float f = (i < 256) ? qo[z * 256 + i] : co[z * 256 + (i - 256)];
    gf[(long long)z * 512 + i] = f * g;
  }
}

// ============ final decode: [max_n gf ; mean_n gf] . dec_W + b, * logics ============
__global__ void decode_k(const float* __restrict__ gf, const float* __restrict__ W,
                         const float* __restrict__ bb, const float* __restrict__ lg,
                         float* __restrict__ out)
{
  int b = blockIdx.x, tid = threadIdx.x;
  __shared__ float red[256];
  float s = 0.0f;
  for (int j = tid; j < 1024; j += 256) {
    float v;
    if (j < 512) {
      float m = -3.402823e38f;
      for (int n = 0; n < 10; ++n) m = fmaxf(m, gf[(long long)(b * 10 + n) * 512 + j]);
      v = m;
    } else {
      float a = 0.0f;
      for (int n = 0; n < 10; ++n) a += gf[(long long)(b * 10 + n) * 512 + (j - 512)];
      v = a * 0.1f;
    }
    s += v * W[j];
  }
  red[tid] = s; __syncthreads();
  for (int st = 128; st > 0; st >>= 1) { if (tid < st) red[tid] += red[tid + st]; __syncthreads(); }
  if (tid == 0) out[b] = (red[0] + bb[0]) * lg[b];
}

// =================================================================
extern "C" void kernel_launch(void* const* d_in, const int* in_sizes, int n_in,
                              void* d_out, int out_size, void* d_ws, size_t ws_size,
                              hipStream_t stream) {
  (void)in_sizes; (void)n_in; (void)out_size; (void)ws_size;
  const int Bn = 10, Nn = 10, Ln = 256, Qn = 100, En = 200, Hn = 128;
  const int BN = Bn * Nn;          // 100
  const int TOKC = BN * Ln;        // 25600
  const int TOKQ = Bn * Qn;        // 1000
  const int KE = 224;              // padded embed dim (200 -> 7*32)
  const int G4 = 4 * Hn;           // 512
  const int D2 = 2 * Hn;           // 256
  const int KQR = 288;             // padded 258 (9*32)
  const int KCR = 544;             // padded 514 (17*32)

  const int*   contents = (const int*)d_in[0];
  const int*   question = (const int*)d_in[1];
  const float* logics   = (const float*)d_in[2];
  const float* emb_f    = (const float*)d_in[3];
  const float* emb_d    = (const float*)d_in[4];
  const float* ctx_Wi   = (const float*)d_in[5];
  const float* ctx_Wh   = (const float*)d_in[6];
  const float* ctx_b    = (const float*)d_in[7];
  const float* qr_Wi    = (const float*)d_in[8];
  const float* qr_Wh    = (const float*)d_in[9];
  const float* qr_b     = (const float*)d_in[10];
  const float* cr_Wi    = (const float*)d_in[11];
  const float* cr_Wh    = (const float*)d_in[12];
  const float* cr_b     = (const float*)d_in[13];
  const float* gate_r_W = (const float*)d_in[14];
  const float* gate_r_b = (const float*)d_in[15];
  const float* gate_n_W = (const float*)d_in[16];
  const float* gate_n_b = (const float*)d_in[17];
  const float* dec_W    = (const float*)d_in[18];
  const float* dec_bb   = (const float*)d_in[19];
  float* out = (float*)d_out;

  // -------- workspace carving (~330 MB) --------
  char* cur = (char*)d_ws;
  auto carve = [&](size_t bytes) { char* p = cur; cur += (bytes + 255) & ~(size_t)255; return p; };
  h16*   wq16     = (h16*)  carve((size_t)TOKQ * KE * 2);
  h16*   wc16     = (h16*)  carve((size_t)TOKC * KE * 2);
  h16*   Wctx_i16 = (h16*)  carve((size_t)2 * KE * G4 * 2);
  h16*   Wctx_h16 = (h16*)  carve((size_t)2 * Hn * G4 * 2);
  h16*   Wqr_i16  = (h16*)  carve((size_t)2 * KQR * G4 * 2);
  h16*   Wqr_h16  = (h16*)  carve((size_t)2 * Hn * G4 * 2);
  h16*   Wcr_i16  = (h16*)  carve((size_t)2 * KCR * G4 * 2);
  h16*   Wcr_h16  = (h16*)  carve((size_t)2 * Hn * G4 * 2);
  float* xg       = (float*)carve((size_t)TOKC * G4 * 4);     // reused by all 4 LSTMs
  float* q_enc    = (float*)carve((size_t)TOKQ * D2 * 4);
  float* D_enc    = (float*)carve((size_t)TOKC * D2 * 4);
  h16*   q_enc16  = (h16*)  carve((size_t)TOKQ * D2 * 2);
  h16*   DRnD16   = (h16*)  carve((size_t)TOKC * 512 * 2);    // [.,0:256]=D_enc, [.,256:512]=RnD
  float* M_qc     = (float*)carve((size_t)BN * Qn * Ln * 4);
  h16*   an16     = (h16*)  carve((size_t)BN * Qn * Ln * 2);
  h16*   bnT16    = (h16*)  carve((size_t)BN * Ln * Qn * 2);  // transposed [z,l,q]
  float* featc    = (float*)carve((size_t)BN * Qn * 2 * 4);
  float* featr    = (float*)carve((size_t)BN * Ln * 2 * 4);
  float* RnQ      = (float*)carve((size_t)BN * Qn * D2 * 4);
  float* gate_q   = (float*)carve((size_t)TOKQ * 4);
  float* gate_c   = (float*)carve((size_t)TOKC * 4);
  float* Mx       = (float*)carve((size_t)2560 * 2560 * 4);   // per-b scratch
  h16*   beta16   = (h16*)  carve((size_t)2560 * 2560 * 2);   // per-b scratch
  float* RmD      = (float*)carve((size_t)TOKC * 512 * 4);
  h16*   rnqin16  = (h16*)  carve((size_t)BN * Qn * KQR * 2);
  h16*   rmdin16  = (h16*)  carve((size_t)BN * Ln * KCR * 2);
  float* qr_hs    = (float*)carve((size_t)BN * Qn * D2 * 4);
  float* cr_hs    = (float*)carve((size_t)BN * Ln * D2 * 4);
  float* q_out    = (float*)carve((size_t)BN * D2 * 4);
  float* c_out    = (float*)carve((size_t)BN * D2 * 4);
  float* gfbuf    = (float*)carve((size_t)BN * 512 * 4);

  auto conv = [&](const float* s, long long sld, h16* dp, long long dld, int sr, int dr, int cols) {
    long long tot = (long long)dr * cols;
    convert2d<<<(int)((tot + 255) / 256), 256, 0, stream>>>(s, sld, dp, dld, sr, dr, cols);
  };
  auto gemm = [&](const h16* A, long long lda, long long sA, int nA,
                  const h16* Bm, long long ldb, long long sB, int nB,
                  float* C, long long ldc, long long sC,
                  h16* Ch, long long ldch, long long sCh,
                  const float* bias, int M, int Ncols, int K, int BT, int batch) {
    dim3 grid((M + 63) / 64, (Ncols + 63) / 64, batch);
    gemm_wmma<<<grid, 128, 0, stream>>>(A, lda, sA, nA, Bm, ldb, sB, nB,
                                        C, ldc, sC, Ch, ldch, sCh, bias, M, Ncols, K, BT);
  };

  // ---- weight conversion to padded f16 ----
  for (int d = 0; d < 2; ++d) {
    conv(ctx_Wi + (long long)d * En * G4,  G4, Wctx_i16 + (long long)d * KE * G4,  G4, En, KE, G4);
    conv(ctx_Wh + (long long)d * Hn * G4,  G4, Wctx_h16 + (long long)d * Hn * G4,  G4, Hn, Hn, G4);
    conv(qr_Wi  + (long long)d * 258 * G4, G4, Wqr_i16  + (long long)d * KQR * G4, G4, 258, KQR, G4);
    conv(qr_Wh  + (long long)d * Hn * G4,  G4, Wqr_h16  + (long long)d * Hn * G4,  G4, Hn, Hn, G4);
    conv(cr_Wi  + (long long)d * 514 * G4, G4, Wcr_i16  + (long long)d * KCR * G4, G4, 514, KCR, G4);
    conv(cr_Wh  + (long long)d * Hn * G4,  G4, Wcr_h16  + (long long)d * Hn * G4,  G4, Hn, Hn, G4);
  }

  // ---- embeddings (fixed + delta) -> padded f16 ----
  gather_embed<<<(TOKQ * KE + 255) / 256, 256, 0, stream>>>(question, TOKQ, emb_f, emb_d, wq16, En, KE);
  gather_embed<<<(TOKC * KE + 255) / 256, 256, 0, stream>>>(contents, TOKC, emb_f, emb_d, wc16, En, KE);

  // ---- context bi-LSTM over contents -> D_enc [25600,256] ----
  for (int d = 0; d < 2; ++d) {
    gemm(wc16, KE, 0, 1, Wctx_i16 + (long long)d * KE * G4, G4, 0, 1,
         xg, G4, 0, nullptr, 0, 0, ctx_b + d * G4, TOKC, G4, KE, 0, 1);
    lstm_scan<<<(BN + 15) / 16, 128, LSTM_SHMEM, stream>>>(
        xg, Wctx_h16 + (long long)d * Hn * G4, D_enc, D2, d * Hn, BN, Ln, d);
  }
  // ---- context bi-LSTM over question -> q_enc [1000,256] ----
  for (int d = 0; d < 2; ++d) {
    gemm(wq16, KE, 0, 1, Wctx_i16 + (long long)d * KE * G4, G4, 0, 1,
         xg, G4, 0, nullptr, 0, 0, ctx_b + d * G4, TOKQ, G4, KE, 0, 1);
    lstm_scan<<<1, 128, LSTM_SHMEM, stream>>>(
        xg, Wctx_h16 + (long long)d * Hn * G4, q_enc, D2, d * Hn, Bn, Qn, d);
  }

  // ---- f16 copies for WMMA consumers ----
  conv(q_enc, D2, q_enc16, D2, TOKQ, TOKQ, D2);
  conv(D_enc, D2, DRnD16, 512, TOKC, TOKC, D2);   // left half of D_RnD

  // ---- reasoning gates ----
  rowdot_sigmoid<<<TOKQ, 128, 0, stream>>>(q_enc, D2, gate_r_W, gate_r_b, gate_q, D2, 1e-8f);
  rowdot_sigmoid<<<TOKC, 128, 0, stream>>>(D_enc, D2, gate_r_W, gate_r_b, gate_c, D2, 1e-8f);

  // ---- M_qc[z] = q_enc16[b] . D_enc16[z]^T   (NT, batched z=100) ----
  gemm(q_enc16, D2, (long long)Qn * D2, Nn,
       DRnD16, 512, (long long)Ln * 512, 1,
       M_qc, Ln, (long long)Qn * Ln, nullptr, 0, 0, nullptr, Qn, Ln, D2, 1, BN);

  // ---- an = softmax over l (+feat_col); bn^T = softmax over q (+feat_row) ----
  softmax_row<<<BN * Qn, 256, 0, stream>>>(M_qc, an16, featc, Ln);
  softmax_col<<<BN * Ln, 128, 0, stream>>>(M_qc, bnT16, featr, Qn, Ln);

  // ---- RnQ[z] = an[z] . D_enc16[z]   (NN) ----
  gemm(an16, Ln, (long long)Qn * Ln, 1,
       DRnD16, 512, (long long)Ln * 512, 1,
       RnQ, D2, (long long)Qn * D2, nullptr, 0, 0, nullptr, Qn, D2, Ln, 0, BN);

  // ---- RnD[z] = bn^T[z] . q_enc16[b]  -> f16 directly into D_RnD right half ----
  gemm(bnT16, Qn, (long long)Ln * Qn, 1,
       q_enc16, D2, (long long)Qn * D2, Nn,
       nullptr, 0, 0, DRnD16 + D2, 512, (long long)Ln * 512,
       nullptr, Ln, D2, Qn, 0, BN);

  // ---- cross-document attention per batch b ----
  for (int b = 0; b < Bn; ++b) {
    const h16* Db = DRnD16 + (long long)b * 2560 * 512;
    gemm(Db, 512, 0, 1, Db, 512, 0, 1, Mx, 2560, 0,
         nullptr, 0, 0, nullptr, 2560, 2560, 512, 1, 1);              // Mx = Db . Db^T
    softmax_row<<<2560, 256, 0, stream>>>(Mx, beta16, nullptr, 2560); // beta
    gemm(beta16, 2560, 0, 1, Db, 512, 0, 1,
         RmD + (long long)b * 2560 * 512, 512, 0,
         nullptr, 0, 0, nullptr, 2560, 512, 2560, 0, 1);              // RmD = beta . Db
  }

  // ---- gated concat inputs for second LSTMs ----
  build_rnq_in<<<(BN * Qn * KQR + 255) / 256, 256, 0, stream>>>(RnQ, featc, gate_q, rnqin16);
  build_rmd_in<<<(BN * Ln * KCR + 255) / 256, 256, 0, stream>>>(RmD, featr, gate_c, rmdin16);

  // ---- qr bi-LSTM (S=100, T=100) ----
  for (int d = 0; d < 2; ++d) {
    gemm(rnqin16, KQR, 0, 1, Wqr_i16 + (long long)d * KQR * G4, G4, 0, 1,
         xg, G4, 0, nullptr, 0, 0, qr_b + d * G4, BN * Qn, G4, KQR, 0, 1);
    lstm_scan<<<(BN + 15) / 16, 128, LSTM_SHMEM, stream>>>(
        xg, Wqr_h16 + (long long)d * Hn * G4, qr_hs, D2, d * Hn, BN, Qn, d);
  }
  // ---- cr bi-LSTM (S=100, T=256) ----
  for (int d = 0; d < 2; ++d) {
    gemm(rmdin16, KCR, 0, 1, Wcr_i16 + (long long)d * KCR * G4, G4, 0, 1,
         xg, G4, 0, nullptr, 0, 0, cr_b + d * G4, BN * Ln, G4, KCR, 0, 1);
    lstm_scan<<<(BN + 15) / 16, 128, LSTM_SHMEM, stream>>>(
        xg, Wcr_h16 + (long long)d * Hn * G4, cr_hs, D2, d * Hn, BN, Ln, d);
  }

  // ---- max over time, node gating, decode ----
  max_time<<<BN, 256, 0, stream>>>(qr_hs, q_out, Qn, D2);
  max_time<<<BN, 256, 0, stream>>>(cr_hs, c_out, Ln, D2);
  gate_feat<<<BN, 128, 0, stream>>>(q_out, c_out, gate_n_W, gate_n_b, gfbuf);
  decode_k<<<Bn, 256, 0, stream>>>(gfbuf, dec_W, dec_bb, logics, out);
}
